// CoreLstm_2534030705145
// MI455X (gfx1250) — matrix operand order repaired and checked by reading, compiled, not run on hardware
//
#include <hip/hip_runtime.h>
#include <hip/hip_bf16.h>
#include <stdint.h>

// Problem constants (from reference)
#define T_STEPS 32
#define BATCH   256
#define HIDN    384
#define FOURH   1536
#define D0      1152
#define NLAYERS 3
#define LN_EPS  1e-5f

typedef __attribute__((ext_vector_type(16))) __bf16 v16bf;
typedef __attribute__((ext_vector_type(8)))  float  v8f;

union FragBF { v16bf v; uint32_t u[8]; };

__device__ __forceinline__ uint16_t f2bf(float f) {
  uint32_t u = __float_as_uint(f);
  u += 0x7FFFu + ((u >> 16) & 1u);   // round-to-nearest-even
  return (uint16_t)(u >> 16);
}

__device__ __forceinline__ float sigf(float x) { return 1.0f / (1.0f + __expf(-x)); }

// ---------------------------------------------------------------------------
// A-fragment: 16x32 bf16 tile of X (row-major [M][Kdim]), per layout:
//   lanes 0-15 : M=lane,  VGPR0-7 K=0..15
//   lanes16-31 : M=l-16,  VGPR0-7 K=16..31
__device__ __forceinline__ v16bf load_a_frag(const uint16_t* X, int row, int Kdim, int k0, int g) {
  FragBF f;
  const uint16_t* p = X + (size_t)row * Kdim + k0 + 16 * g;
#pragma unroll
  for (int j = 0; j < 8; ++j)
    f.u[j] = *(const uint32_t*)(p + 2 * j);
  return f.v;
}

// B-fragment: 32x16 bf16 tile of W^T (stored [N][Kdim] row-major), per layout:
//   VGPR j: lanes 0-15 hold K=2j,2j+1 (N=lane); lanes 16-31 hold K=16+2j,+1
__device__ __forceinline__ v16bf load_b_frag(const uint16_t* Wt, int col, int Kdim, int k0, int g) {
  FragBF f;
  const uint16_t* p = Wt + (size_t)col * Kdim + k0 + 16 * g;
#pragma unroll
  for (int j = 0; j < 8; ++j)
    f.u[j] = *(const uint32_t*)(p + 2 * j);
  return f.v;
}

// ---------------------------------------------------------------------------
// Generic bf16 WMMA GEMM: C = A[M][Kdim] * Bt[N][Kdim]^T, fp32 accumulate.
// Each wave computes a (16*MT) x (16*NT) tile; A fragments reused across NT
// WMMAs, B fragments reused across MT WMMAs.
template <int MT, int NT>
__global__ void gemm_bf16_wmma(const uint16_t* __restrict__ A,
                               const uint16_t* __restrict__ Bt,
                               float* __restrict__ C,
                               int M, int N, int Kdim) {
  const int wave = blockIdx.x * (blockDim.x >> 5) + (threadIdx.x >> 5);
  const int lane = threadIdx.x & 31;
  const int tilesN = N / (16 * NT);
  const int totalWaves = (M / (16 * MT)) * tilesN;
  if (wave >= totalWaves) return;
  const int tm = wave / tilesN;
  const int tn = wave % tilesN;
  const int g = lane >> 4;
  const int r = lane & 15;
  const int m0 = tm * 16 * MT;
  const int n0 = tn * 16 * NT;

  v8f acc[MT][NT];
#pragma unroll
  for (int i = 0; i < MT; ++i)
#pragma unroll
    for (int j = 0; j < NT; ++j) acc[i][j] = (v8f){};

  for (int k0 = 0; k0 < Kdim; k0 += 32) {
    // Prefetch next K-slice (gfx1250 global_prefetch; L2-hot weights, cheap hint)
    if (k0 + 32 < Kdim) {
      __builtin_prefetch(A + (size_t)(m0 + r) * Kdim + k0 + 32, 0, 3);
      __builtin_prefetch(Bt + (size_t)(n0 + r) * Kdim + k0 + 32, 0, 3);
    }
    v16bf a[MT];
#pragma unroll
    for (int i = 0; i < MT; ++i)
      a[i] = load_a_frag(A, m0 + i * 16 + r, Kdim, k0, g);
#pragma unroll
    for (int j = 0; j < NT; ++j) {
      v16bf b = load_b_frag(Bt, n0 + j * 16 + r, Kdim, k0, g);
#pragma unroll
      for (int i = 0; i < MT; ++i)
        acc[i][j] = __builtin_amdgcn_wmma_f32_16x16x32_bf16(
            false, a[i], false, b, (short)0, acc[i][j], false, false);
    }
  }
  // C/D layout: lanes 0-15: M=m0+q, N=lane; lanes 16-31: M=m0+8+q
#pragma unroll
  for (int i = 0; i < MT; ++i) {
#pragma unroll
    for (int j = 0; j < NT; ++j) {
      const int n = n0 + j * 16 + r;
#pragma unroll
      for (int q = 0; q < 8; ++q) {
        const int m = m0 + i * 16 + q + 8 * g;
        C[(size_t)m * N + n] = acc[i][j][q];
      }
    }
  }
}

// ---------------------------------------------------------------------------
// Concatenate (entity|spatial|scalar) -> bf16 X [T*B][1152]
__global__ void prep_concat(const float* __restrict__ ent,
                            const float* __restrict__ spa,
                            const float* __restrict__ sca,
                            uint16_t* __restrict__ X) {
  size_t idx = (size_t)blockIdx.x * blockDim.x + threadIdx.x;
  const size_t total = (size_t)T_STEPS * BATCH * D0;
  if (idx >= total) return;
  int k = (int)(idx % D0);
  size_t tb = idx / D0;
  float v;
  if (k < 256)       v = ent[tb * 256 + k];
  else if (k < 512)  v = spa[tb * 256 + (k - 256)];
  else               v = sca[tb * 640 + (k - 512)];
  X[idx] = f2bf(v);
}

// W [Kdim][N] fp32 -> Wt [N][Kdim] bf16 (so K is contiguous for B-fragments)
__global__ void transpose_w(const float* __restrict__ W, uint16_t* __restrict__ Wt,
                            int Kdim, int N) {
  size_t idx = (size_t)blockIdx.x * blockDim.x + threadIdx.x;
  size_t total = (size_t)Kdim * N;
  if (idx >= total) return;
  int k = (int)(idx % Kdim);
  int n = (int)(idx / Kdim);
  Wt[idx] = f2bf(W[(size_t)k * N + n]);
}

// Row LayerNorm over N=1536, then *gamma + beta (+ lstm bias), in place.
__global__ void ln_rows(float* __restrict__ X,
                        const float* __restrict__ gamma,
                        const float* __restrict__ beta,
                        const float* __restrict__ bias) {
  const int N = FOURH;
  float* x = X + (size_t)blockIdx.x * N;
  const int tid = threadIdx.x;
  __shared__ float sh[512];
  float s = 0.f, q = 0.f;
  for (int j = tid; j < N; j += 256) { float v = x[j]; s += v; q += v * v; }
  sh[tid] = s; sh[256 + tid] = q;
  __syncthreads();
  for (int off = 128; off > 0; off >>= 1) {
    if (tid < off) { sh[tid] += sh[tid + off]; sh[256 + tid] += sh[256 + tid + off]; }
    __syncthreads();
  }
  const float mu = sh[0] / N;
  const float var = sh[256] / N - mu * mu;
  const float rstd = rsqrtf(var + LN_EPS);
  for (int j = tid; j < N; j += 256)
    x[j] = (x[j] - mu) * rstd * gamma[j] + beta[j] + bias[j];
}

// Initialize working h (bf16) and c (fp32) from h0/c0 of this layer.
__global__ void init_hc(const float* __restrict__ h0l, const float* __restrict__ c0l,
                        uint16_t* __restrict__ hbf, float* __restrict__ c) {
  int i = blockIdx.x * blockDim.x + threadIdx.x;
  if (i >= BATCH * HIDN) return;
  hbf[i] = f2bf(h0l[i]);
  c[i] = c0l[i];
}

// Fused: LN(h@Wh) + xg_t -> gates -> c,h update. One block per batch row.
__global__ void gate_step(const float* __restrict__ G,      // [B][1536] = h@Wh
                          const float* __restrict__ XGt,    // [B][1536] xg at time t
                          const float* __restrict__ ghw,
                          const float* __restrict__ ghb,
                          float* __restrict__ c,            // [B][384] in/out
                          uint16_t* __restrict__ hbf,       // [B][384] out (next-step A)
                          uint16_t* __restrict__ xnext,     // optional bf16 seq for next layer
                          float* __restrict__ yout,         // optional fp32 output seq
                          float* __restrict__ hn,           // optional final H
                          float* __restrict__ cn) {         // optional final C
  const int b = blockIdx.x;
  const int tid = threadIdx.x;
  const float* g = G + (size_t)b * FOURH;
  const float* xg = XGt + (size_t)b * FOURH;

  __shared__ float sh[512];
  float s = 0.f, q = 0.f;
  for (int j = tid; j < FOURH; j += 256) { float v = g[j]; s += v; q += v * v; }
  sh[tid] = s; sh[256 + tid] = q;
  __syncthreads();
  for (int off = 128; off > 0; off >>= 1) {
    if (tid < off) { sh[tid] += sh[tid + off]; sh[256 + tid] += sh[256 + tid + off]; }
    __syncthreads();
  }
  const float mu = sh[0] / FOURH;
  const float var = sh[256] / FOURH - mu * mu;
  const float rstd = rsqrtf(var + LN_EPS);

  for (int j = tid; j < HIDN; j += 256) {
    float gi = xg[j]            + (g[j]            - mu) * rstd * ghw[j]            + ghb[j];
    float gf = xg[HIDN + j]     + (g[HIDN + j]     - mu) * rstd * ghw[HIDN + j]     + ghb[HIDN + j];
    float go = xg[2 * HIDN + j] + (g[2 * HIDN + j] - mu) * rstd * ghw[2 * HIDN + j] + ghb[2 * HIDN + j];
    float gu = xg[3 * HIDN + j] + (g[3 * HIDN + j] - mu) * rstd * ghw[3 * HIDN + j] + ghb[3 * HIDN + j];
    float cc = sigf(gf) * c[(size_t)b * HIDN + j] + sigf(gi) * tanhf(gu);
    float hh = sigf(go) * tanhf(cc);
    c[(size_t)b * HIDN + j] = cc;
    hbf[(size_t)b * HIDN + j] = f2bf(hh);
    if (xnext) xnext[(size_t)b * HIDN + j] = f2bf(hh);
    if (yout)  yout[(size_t)b * HIDN + j] = hh;
    if (hn)    hn[(size_t)b * HIDN + j] = hh;
    if (cn)    cn[(size_t)b * HIDN + j] = cc;
  }
}

// ---------------------------------------------------------------------------
extern "C" void kernel_launch(void* const* d_in, const int* in_sizes, int n_in,
                              void* d_out, int out_size, void* d_ws, size_t ws_size,
                              hipStream_t stream) {
  const float* ent  = (const float*)d_in[0];
  const float* spa  = (const float*)d_in[1];
  const float* sca  = (const float*)d_in[2];
  const float* h0   = (const float*)d_in[3];
  const float* c0   = (const float*)d_in[4];
  const float* wx0  = (const float*)d_in[5];
  const float* wx1  = (const float*)d_in[6];
  const float* wx2  = (const float*)d_in[7];
  const float* wh   = (const float*)d_in[8];
  const float* bias = (const float*)d_in[9];
  const float* gxw  = (const float*)d_in[10];
  const float* gxb  = (const float*)d_in[11];
  const float* ghw  = (const float*)d_in[12];
  const float* ghb  = (const float*)d_in[13];
  float* out = (float*)d_out;

  const size_t ROWS = (size_t)T_STEPS * BATCH;   // 8192
  char* ws = (char*)d_ws;
  size_t off = 0;
  auto carve = [&](size_t bytes) { void* p = ws + off; off += (bytes + 255) & ~(size_t)255; return p; };
  uint16_t* XA   = (uint16_t*)carve(ROWS * D0 * 2);              // layer0 input (also layer2 input, 384-wide)
  uint16_t* XB   = (uint16_t*)carve(ROWS * HIDN * 2);            // layer1 input
  uint16_t* WX0T = (uint16_t*)carve((size_t)D0 * FOURH * 2);
  uint16_t* WX1T = (uint16_t*)carve((size_t)HIDN * FOURH * 2);
  uint16_t* WX2T = (uint16_t*)carve((size_t)HIDN * FOURH * 2);
  uint16_t* WHT  = (uint16_t*)carve((size_t)NLAYERS * HIDN * FOURH * 2);
  float*    XG   = (float*)carve(ROWS * FOURH * 4);
  float*    G    = (float*)carve((size_t)BATCH * FOURH * 4);
  float*    Cbuf = (float*)carve((size_t)BATCH * HIDN * 4);
  uint16_t* HBF  = (uint16_t*)carve((size_t)BATCH * HIDN * 2);
  (void)ws_size; (void)in_sizes; (void)n_in; (void)out_size;

  // Output regions
  const size_t OUT_Y = 0;
  const size_t OUT_H = (size_t)T_STEPS * BATCH * HIDN;           // 3145728
  const size_t OUT_C = OUT_H + (size_t)NLAYERS * BATCH * HIDN;   // + 294912

  // 1) Concat inputs -> bf16
  {
    size_t tot = ROWS * D0;
    prep_concat<<<(unsigned)((tot + 255) / 256), 256, 0, stream>>>(ent, spa, sca, XA);
  }
  // 2) Transpose+convert weights -> bf16 [N][K]
  {
    size_t t0 = (size_t)D0 * FOURH;
    transpose_w<<<(unsigned)((t0 + 255) / 256), 256, 0, stream>>>(wx0, WX0T, D0, FOURH);
    size_t t1 = (size_t)HIDN * FOURH;
    transpose_w<<<(unsigned)((t1 + 255) / 256), 256, 0, stream>>>(wx1, WX1T, HIDN, FOURH);
    transpose_w<<<(unsigned)((t1 + 255) / 256), 256, 0, stream>>>(wx2, WX2T, HIDN, FOURH);
    for (int l = 0; l < NLAYERS; ++l)
      transpose_w<<<(unsigned)((t1 + 255) / 256), 256, 0, stream>>>(
          wh + (size_t)l * HIDN * FOURH, WHT + (size_t)l * HIDN * FOURH, HIDN, FOURH);
  }

  for (int l = 0; l < NLAYERS; ++l) {
    const int Kdim = (l == 0) ? D0 : HIDN;
    const uint16_t* Xin = (l == 0) ? XA : (l == 1 ? XB : XA);
    const uint16_t* WxT = (l == 0) ? WX0T : (l == 1 ? WX1T : WX2T);

    // Input projection over all timesteps: XG = Xin @ Wx  (32x64 tiles/wave)
    {
      int waves = (int)((ROWS / 32) * (FOURH / 64));   // 6144
      int blocks = (waves + 7) / 8;                    // 768
      gemm_bf16_wmma<2, 4><<<blocks, 256, 0, stream>>>(Xin, WxT, XG, (int)ROWS, FOURH, Kdim);
    }
    // Row LN + gamma/beta + lstm bias folded in
    ln_rows<<<(unsigned)ROWS, 256, 0, stream>>>(XG, gxw + (size_t)l * FOURH,
                                                gxb + (size_t)l * FOURH,
                                                bias + (size_t)l * FOURH);
    // Init h (bf16) and c
    init_hc<<<(BATCH * HIDN + 255) / 256, 256, 0, stream>>>(
        h0 + (size_t)l * BATCH * HIDN, c0 + (size_t)l * BATCH * HIDN, HBF, Cbuf);

    uint16_t* xnext = (l == 0) ? XB : (l == 1 ? XA : nullptr);
    const uint16_t* WhTl = WHT + (size_t)l * HIDN * FOURH;

    for (int t = 0; t < T_STEPS; ++t) {
      // Recurrent GEMM: G = h_bf @ Wh  (16x32 tiles/wave -> 768 waves for latency)
      {
        int waves = (BATCH / 16) * (FOURH / 32);   // 768
        int blocks = (waves + 7) / 8;              // 96
        gemm_bf16_wmma<1, 2><<<blocks, 256, 0, stream>>>(HBF, WhTl, G, BATCH, FOURH, HIDN);
      }
      float* yout = (l == NLAYERS - 1) ? (out + OUT_Y + (size_t)t * BATCH * HIDN) : nullptr;
      float* hn = (t == T_STEPS - 1) ? (out + OUT_H + (size_t)l * BATCH * HIDN) : nullptr;
      float* cn = (t == T_STEPS - 1) ? (out + OUT_C + (size_t)l * BATCH * HIDN) : nullptr;
      uint16_t* xn = xnext ? (xnext + (size_t)t * BATCH * HIDN) : nullptr;
      gate_step<<<BATCH, 256, 0, stream>>>(G, XG + (size_t)t * BATCH * FOURH,
                                           ghw + (size_t)l * FOURH, ghb + (size_t)l * FOURH,
                                           Cbuf, HBF, xn, yout, hn, cn);
    }
  }
}
